// LlamaMLP_32916629357390
// MI455X (gfx1250) — matrix operand order repaired and checked by reading, not compiled, not run on hardware
//
#include <hip/hip_runtime.h>
#include <cstdint>

// ---------------------------------------------------------------------------
// Problem constants (from reference setup_inputs)
// ---------------------------------------------------------------------------
constexpr int BB = 2;
constexpr int S  = 2048;
constexpr int H  = 2048;
constexpr int I  = 8192;
constexpr int N  = BB * S;           // 4096 tokens

// ---------------------------------------------------------------------------
// Vector types
// ---------------------------------------------------------------------------
typedef __attribute__((ext_vector_type(16))) __bf16        v16bf;
typedef __attribute__((ext_vector_type(32))) __bf16        v32bf;
typedef __attribute__((ext_vector_type(8)))  float         v8f;
typedef __attribute__((ext_vector_type(8)))  unsigned int  u32x8;
typedef __attribute__((ext_vector_type(16))) unsigned int  u32x16;
typedef __attribute__((ext_vector_type(4)))  unsigned int  u32x4;
typedef __attribute__((ext_vector_type(8)))  int           i32x8;
typedef __attribute__((ext_vector_type(4)))  int           i32x4;

#if __has_builtin(__builtin_amdgcn_tensor_load_to_lds) && __has_builtin(__builtin_amdgcn_s_wait_tensorcnt)
#define HAVE_TDM 1
#else
#define HAVE_TDM 0
#endif

// round-to-nearest-even f32 -> bf16 bit pattern
__device__ __forceinline__ unsigned short f2bf(float f) {
    union { float f; unsigned u; } x; x.f = f;
    unsigned u = x.u;
    unsigned r = (u + 0x7FFFu + ((u >> 16) & 1u)) >> 16;
    return (unsigned short)r;
}

__device__ __forceinline__ unsigned lds_off(const void* p) {
    // generic LDS address: low 32 bits are the LDS byte offset
    return (unsigned)(unsigned long long)p;
}

#if HAVE_TDM
// ---------------------------------------------------------------------------
// TDM 2D tile load: rows of tile_d0 bf16 elements, tile_d1 rows, row stride
// 'stride_elems' in global memory; LDS rows padded via pad_interval/pad_amount.
// D# bit layout per CDNA5 ISA ch.8 (data_size=1 => 2-byte elements).
// 6-arg builtin form (clang-23 / therock headers toolchain).
// ---------------------------------------------------------------------------
__device__ __forceinline__ void tdm_load_2d(unsigned lds_byte_off, const void* gaddr,
                                            unsigned tile_d0, unsigned tile_d1,
                                            unsigned stride_elems,
                                            unsigned pad_interval_code,
                                            unsigned pad_amount_code) {
    unsigned long long ga = (unsigned long long)gaddr;
    u32x4 g0;
    g0[0] = 1u;                                            // count=1, user desc
    g0[1] = lds_byte_off;                                  // lds_addr
    g0[2] = (unsigned)(ga & 0xFFFFFFFFu);                  // global_addr lo
    g0[3] = (unsigned)((ga >> 32) & 0x1FFFFFFu)            // global_addr hi (57b)
          | (2u << 30);                                    // type = 2 ("image")
    const unsigned td0 = 0x40000000u, td1 = 0x40000000u;   // huge dims: no OOB
    i32x8 g1;
    g1[0] = (int)((1u << 16)                               // data_size = 2B
                | (1u << 20)                               // pad_enable
                | (pad_interval_code << 22)
                | (pad_amount_code << 25));
    g1[1] = (int)((td0 & 0xFFFFu) << 16);                  // tensor_dim0 lo16
    g1[2] = (int)(((td0 >> 16) & 0xFFFFu) | ((td1 & 0xFFFFu) << 16));
    g1[3] = (int)(((td1 >> 16) & 0xFFFFu) | ((tile_d0 & 0xFFFFu) << 16));
    g1[4] = (int)(tile_d1 & 0xFFFFu);                      // tile_dim1; tile_dim2=0
    g1[5] = (int)stride_elems;                             // tensor_dim0_stride lo32
    g1[6] = 0;                                             // stride hi16, dim1_stride
    g1[7] = 0;
    i32x4 z4 = {0, 0, 0, 0};
    i32x8 z8 = {0, 0, 0, 0, 0, 0, 0, 0};
    __builtin_amdgcn_tensor_load_to_lds(g0, g1, z4, z4, z8, 0);
}
#endif

// ---------------------------------------------------------------------------
// Kernel 0a: permute tokens + convert x f32 -> bf16    xpb[N, H]
// ---------------------------------------------------------------------------
__global__ void k_cvt_x(const float* __restrict__ x, const int* __restrict__ perm,
                        unsigned short* __restrict__ xpb) {
    long tid  = (long)blockIdx.x * 256 + threadIdx.x;
    long base = tid * 8;
    int  row  = (int)(base / H);
    int  col  = (int)(base % H);
    int  b    = row / S, jl = row % S;
    long src  = (long)(b * S + perm[jl]) * H + col;
    const float4* pin = (const float4*)(x + src);
    float4 f0 = pin[0], f1 = pin[1];
    unsigned short o[8];
    o[0] = f2bf(f0.x); o[1] = f2bf(f0.y); o[2] = f2bf(f0.z); o[3] = f2bf(f0.w);
    o[4] = f2bf(f1.x); o[5] = f2bf(f1.y); o[6] = f2bf(f1.z); o[7] = f2bf(f1.w);
    *(uint4*)(xpb + base) = *(const uint4*)o;
}

// ---------------------------------------------------------------------------
// Kernel 0b: f32 -> bf16 weight conversion (layout preserved)
// ---------------------------------------------------------------------------
__global__ void k_cvt(const float* __restrict__ in, unsigned short* __restrict__ out) {
    long base = ((long)blockIdx.x * 256 + threadIdx.x) * 8;
    const float4* pin = (const float4*)(in + base);
    float4 f0 = pin[0], f1 = pin[1];
    unsigned short o[8];
    o[0] = f2bf(f0.x); o[1] = f2bf(f0.y); o[2] = f2bf(f0.z); o[3] = f2bf(f0.w);
    o[4] = f2bf(f1.x); o[5] = f2bf(f1.y); o[6] = f2bf(f1.z); o[7] = f2bf(f1.w);
    *(uint4*)(out + base) = *(const uint4*)o;
}

// ---------------------------------------------------------------------------
// Kernel 1: GEMM1 (y1 = xp @ w1^T) + squared ReLU + 2:4 compress
//   Block 256 thr = 8 waves; tile 128 tokens x 64 I-cols; K-step 64 (2 WMMA/frag).
//   B tiles: TDM double-buffered into LDS (row stride 64+8 bf16 via TDM pad).
// ---------------------------------------------------------------------------
__global__ void __launch_bounds__(256, 1)
k_gemm1(const unsigned short* __restrict__ xpb, const unsigned short* __restrict__ w1b,
        unsigned short* __restrict__ y2p, unsigned int* __restrict__ y2idx) {
    __shared__ unsigned short Bs[2][64 * 72];     // 64 rows x (64 K + 8 pad) bf16
    __shared__ float Ys[8][16][68];               // per-wave 16x64 f32 tile

    const int tid  = threadIdx.x;
    const int lane = tid & 31;
    const int wv   = tid >> 5;
    const int half = lane >> 4;
    const int l16  = lane & 15;
    const int t0   = blockIdx.x * 128 + wv * 16;
    const int iBase = blockIdx.y * 64;
    constexpr int NK = H / 64;                    // 32 staged tiles

    v8f c[4] = {v8f{}, v8f{}, v8f{}, v8f{}};

#if HAVE_TDM
    if (wv == 0)                                  // prologue: stage tile 0
        tdm_load_2d(lds_off(&Bs[0][0]), w1b + (long)iBase * H,
                    /*d0*/64, /*d1*/64, /*stride*/H, /*int:128B*/4, /*pad:16B*/3);
#endif

    for (int kb = 0; kb < NK; ++kb) {
#if HAVE_TDM
        if (wv == 0) __builtin_amdgcn_s_wait_tensorcnt(0);
        __syncthreads();                          // tile kb ready for all waves
        if (wv == 0 && kb + 1 < NK)               // overlap DMA of tile kb+1
            tdm_load_2d(lds_off(&Bs[(kb + 1) & 1][0]),
                        w1b + (long)iBase * H + (kb + 1) * 64,
                        64, 64, H, 4, 3);
        const unsigned short* Bbuf = &Bs[kb & 1][0];
#else
        __syncthreads();
        {
            int il = tid >> 2, kc = (tid & 3) * 16;
            const uint4* src = (const uint4*)(w1b + (long)(iBase + il) * H + kb * 64 + kc);
            uint4 d0 = src[0], d1 = src[1];
            uint4* dst = (uint4*)&Bs[0][il * 72 + kc];
            dst[0] = d0; dst[1] = d1;
        }
        __syncthreads();
        const unsigned short* Bbuf = &Bs[0][0];
#endif
        // two K=32 WMMA sub-steps per staged tile
#pragma unroll
        for (int s = 0; s < 2; ++s) {
            const unsigned short* ar = xpb + (long)(t0 + l16) * H + kb * 64 + s * 32;
            uint4 alo = *(const uint4*)(ar + 8 * half);
            uint4 ahi = *(const uint4*)(ar + 16 + 8 * half);
            u32x8 au = {alo.x, alo.y, alo.z, alo.w, ahi.x, ahi.y, ahi.z, ahi.w};
            v16bf a  = __builtin_bit_cast(v16bf, au);
#pragma unroll
            for (int j = 0; j < 4; ++j) {
                const unsigned short* bp =
                    Bbuf + (16 * j + l16) * 72 + s * 32 + 8 * half;
                uint4 b0 = *(const uint4*)bp, b1 = *(const uint4*)(bp + 16);
                u32x8 bu = {b0.x, b0.y, b0.z, b0.w, b1.x, b1.y, b1.z, b1.w};
                v16bf bf = __builtin_bit_cast(v16bf, bu);
                c[j] = __builtin_amdgcn_wmma_f32_16x16x32_bf16(
                    false, a, false, bf, (short)0, c[j], false, false);
            }
        }
    }

    // ---- accumulators -> LDS ----
#pragma unroll
    for (int j = 0; j < 4; ++j)
#pragma unroll
        for (int r = 0; r < 8; ++r)
            Ys[wv][r + 8 * half][16 * j + l16] = c[j][r];

    // ---- squared ReLU + 2:4 compress: lane = (row l16, 32-col chunk half) ----
    const float* yr = &Ys[wv][l16][32 * half];
    unsigned int word = 0;
    unsigned short packed[16];
#pragma unroll
    for (int g = 0; g < 8; ++g) {
        float v[4];
#pragma unroll
        for (int q = 0; q < 4; ++q) {
            float f = yr[4 * g + q];
            v[q] = f > 0.f ? f * f : 0.f;
        }
        int i0 = 0; float m0 = v[0];
#pragma unroll
        for (int q = 1; q < 4; ++q) if (v[q] > m0) { m0 = v[q]; i0 = q; }
        int i1 = -1; float m1 = -1.f;
#pragma unroll
        for (int q = 0; q < 4; ++q) if (q != i0 && v[q] > m1) { m1 = v[q]; i1 = q; }
        int lo = i0 < i1 ? i0 : i1;
        int hi = i0 < i1 ? i1 : i0;
        word |= (unsigned)(lo | (hi << 2)) << (4 * g);
        packed[2 * g]     = f2bf(v[lo]);
        packed[2 * g + 1] = f2bf(v[hi]);
    }

    const int token = t0 + l16;
    long pbase = (long)token * (I / 2) + (iBase + 32 * half) / 2;
    *(uint4*)(y2p + pbase)     = ((const uint4*)packed)[0];
    *(uint4*)(y2p + pbase + 8) = ((const uint4*)packed)[1];
    y2idx[(long)token * (I / 32) + (iBase / 32 + half)] = word;
}

// ---------------------------------------------------------------------------
// Kernel 2: sparse GEMM2 (y3 = y2s @ w2^T) via V_SWMMAC_F32_16X16X64_BF16
//   Block 256 thr = 8 waves; tile 128 tokens x 64 H-cols; K-step 128 (2 SWMMAC).
//   B tiles: TDM double-buffered (row stride 128+8 bf16 via TDM pad).
// ---------------------------------------------------------------------------
__global__ void __launch_bounds__(256, 1)
k_gemm2(const unsigned short* __restrict__ y2p, const unsigned int* __restrict__ y2idx,
        const unsigned short* __restrict__ w2b, const int* __restrict__ perm,
        float* __restrict__ out) {
    __shared__ unsigned short Bs[2][64 * 136];    // 64 rows x (128 K + 8 pad) bf16

    const int tid  = threadIdx.x;
    const int lane = tid & 31;
    const int wv   = tid >> 5;
    const int half = lane >> 4;
    const int l16  = lane & 15;
    const int t0   = blockIdx.x * 128 + wv * 16;
    const int h0   = blockIdx.y * 64;
    constexpr int NK = I / 128;                   // 64 staged tiles

    v8f c[4] = {v8f{}, v8f{}, v8f{}, v8f{}};

#if HAVE_TDM
    if (wv == 0)
        tdm_load_2d(lds_off(&Bs[0][0]), w2b + (long)h0 * I,
                    /*d0*/128, /*d1*/64, /*stride*/I, /*int:256B*/5, /*pad:16B*/3);
#endif

    for (int kb = 0; kb < NK; ++kb) {
#if HAVE_TDM
        if (wv == 0) __builtin_amdgcn_s_wait_tensorcnt(0);
        __syncthreads();
        if (wv == 0 && kb + 1 < NK)
            tdm_load_2d(lds_off(&Bs[(kb + 1) & 1][0]),
                        w2b + (long)h0 * I + (kb + 1) * 128,
                        128, 64, I, 5, 3);
        const unsigned short* Bbuf = &Bs[kb & 1][0];
#else
        __syncthreads();
        {
            int hl = tid >> 2, kc = (tid & 3) * 16;
#pragma unroll
            for (int rep = 0; rep < 2; ++rep) {
                const uint4* src =
                    (const uint4*)(w2b + (long)(h0 + hl) * I + kb * 128 + rep * 64 + kc);
                uint4 d0 = src[0], d1 = src[1];
                uint4* dst = (uint4*)&Bs[0][hl * 136 + rep * 64 + kc];
                dst[0] = d0; dst[1] = d1;
            }
        }
        __syncthreads();
        const unsigned short* Bbuf = &Bs[0][0];
#endif
        // two K=64 SWMMAC sub-steps per staged tile
#pragma unroll
        for (int s = 0; s < 2; ++s) {
            // packed A (16x32 bf16, dense 16-bit A lane layout)
            const unsigned short* ar =
                y2p + (long)(t0 + l16) * (I / 2) + kb * 64 + s * 32;
            uint4 alo = *(const uint4*)(ar + 8 * half);
            uint4 ahi = *(const uint4*)(ar + 16 + 8 * half);
            u32x8 au = {alo.x, alo.y, alo.z, alo.w, ahi.x, ahi.y, ahi.z, ahi.w};
            v16bf a  = __builtin_bit_cast(v16bf, au);
            // sparse index word: lane<16 covers K 0..31 of this 64-K step
            int idxw = (int)y2idx[(long)(t0 + l16) * (I / 32) + kb * 4 + s * 2 + half];
#pragma unroll
            for (int j = 0; j < 4; ++j) {
                const unsigned short* brow = Bbuf + (16 * j + l16) * 136 + s * 64;
                const uint4* bp0 = (const uint4*)(brow + 16 * half);
                const uint4* bp1 = (const uint4*)(brow + 32 + 16 * half);
                uint4 b0 = bp0[0], b1 = bp0[1], b2 = bp1[0], b3 = bp1[1];
                u32x16 bu = {b0.x, b0.y, b0.z, b0.w, b1.x, b1.y, b1.z, b1.w,
                             b2.x, b2.y, b2.z, b2.w, b3.x, b3.y, b3.z, b3.w};
                v32bf bf = __builtin_bit_cast(v32bf, bu);
                c[j] = __builtin_amdgcn_swmmac_f32_16x16x64_bf16(
                    false, a, false, bf, c[j], idxw, false, false);
            }
        }
    }

    // ---- scatter output through inverse permutation: out[b, perm[jl], h] ----
    int srow[8];
#pragma unroll
    for (int r = 0; r < 8; ++r) {
        int token = t0 + r + 8 * half;
        int b = token / S, jl = token % S;
        srow[r] = b * S + perm[jl];
    }
#pragma unroll
    for (int j = 0; j < 4; ++j)
#pragma unroll
        for (int r = 0; r < 8; ++r)
            out[(long)srow[r] * H + h0 + 16 * j + l16] = c[j][r];
}

// ---------------------------------------------------------------------------
// Launcher
// ---------------------------------------------------------------------------
extern "C" void kernel_launch(void* const* d_in, const int* in_sizes, int n_in,
                              void* d_out, int out_size, void* d_ws, size_t ws_size,
                              hipStream_t stream) {
    const float* x    = (const float*)d_in[0];   // [B,S,H]
    const float* w1   = (const float*)d_in[1];   // [I,H]
    const float* w2   = (const float*)d_in[2];   // [H,I]
    const int*   perm = (const int*)d_in[3];     // [S]
    float*       out  = (float*)d_out;           // [B,S,H]

    unsigned short* xpb = (unsigned short*)d_ws;                       // N*H
    unsigned short* w1b = xpb + (size_t)N * H;                         // I*H
    unsigned short* w2b = w1b + (size_t)I * H;                         // H*I
    unsigned short* y2p = w2b + (size_t)H * I;                         // N*(I/2)
    unsigned int*   y2x = (unsigned int*)(y2p + (size_t)N * (I / 2));  // N*(I/32)

    k_cvt_x<<<(N * (long)H) / 2048, 256, 0, stream>>>(x, perm, xpb);
    k_cvt  <<<((long)I * H) / 2048, 256, 0, stream>>>(w1, w1b);
    k_cvt  <<<((long)H * I) / 2048, 256, 0, stream>>>(w2, w2b);

    dim3 g1(N / 128, I / 64);
    k_gemm1<<<g1, 256, 0, stream>>>(xpb, w1b, y2p, y2x);

    dim3 g2(N / 128, H / 64);
    k_gemm2<<<g2, 256, 0, stream>>>(y2p, y2x, w2b, perm, out);

    (void)in_sizes; (void)n_in; (void)out_size; (void)ws_size;
}
